// DifferentialAttentionLayer_74577812127947
// MI455X (gfx1250) — matrix operand order repaired and checked by reading, not compiled, not run on hardware
//
#include <hip/hip_runtime.h>
#include <cstdint>
#include <cstddef>

#define D_MODEL 1024
#define N_HEADS 16
#define POOL    32
#define D_K     64
#define BATCH   32
#define SEQ     512
#define TEMP    0.5f

typedef __attribute__((ext_vector_type(16))) __bf16 bf16x16;
typedef __attribute__((ext_vector_type(8)))  float  f32x8;
typedef unsigned short u16;
typedef unsigned int   u32;

union FragAB { bf16x16 v; uint4 q[2]; u16 s[16]; };

__device__ __forceinline__ u16 f2bf(float f) {
    u32 u = __builtin_bit_cast(u32, f);
    u32 r = u + 0x7FFFu + ((u >> 16) & 1u);
    return (u16)(r >> 16);
}

__device__ __forceinline__ u32 pack2bf(float a, float b) {
    return (u32)f2bf(a) | ((u32)f2bf(b) << 16);
}

__device__ __forceinline__ f32x8 wmma_bf16(bf16x16 a, bf16x16 b, f32x8 c) {
    return __builtin_amdgcn_wmma_f32_16x16x32_bf16(false, a, false, b, (short)0, c, false, false);
}

// A-matrix 16x32 bf16 fragment: lane<16 -> row=lane, K runs [0..7],[16..23];
// lane>=16 -> row=lane-16, K runs [8..15],[24..31].
__device__ __forceinline__ bf16x16 load_a(const u16* base, int stride, int lane) {
    int r  = lane & 15;
    int ko = (lane >> 4) * 8;
    FragAB f;
    f.q[0] = *(const uint4*)(base + (size_t)r * stride + ko);
    f.q[1] = *(const uint4*)(base + (size_t)r * stride + ko + 16);
    return f.v;
}

// B-matrix 32x16 bf16 fragment, K contiguous in memory per column:
// lane<16 -> col=lane, K runs [0..7],[16..23]; lane>=16 -> col=lane-16, K runs [8..15],[24..31].
__device__ __forceinline__ bf16x16 load_b_kcontig(const u16* base, int nstride, int lane) {
    int n  = lane & 15;
    int ko = (lane >> 4) * 8;
    FragAB f;
    f.q[0] = *(const uint4*)(base + (size_t)n * nstride + ko);
    f.q[1] = *(const uint4*)(base + (size_t)n * nstride + ko + 16);
    return f.v;
}

// ---------------------------------------------------------------- converters
__global__ void k_f32_to_bf16(const float* __restrict__ src, u16* __restrict__ dst, int n) {
    int i  = blockIdx.x * blockDim.x + threadIdx.x;
    int st = gridDim.x * blockDim.x;
    for (; i < n; i += st) dst[i] = f2bf(src[i]);
}

// [POOL][D_MODEL][D_K] f32 -> [POOL][D_K][D_MODEL] bf16 (transpose inner 2 dims)
__global__ void k_conv_w_t(const float* __restrict__ src, u16* __restrict__ dst) {
    int i  = blockIdx.x * blockDim.x + threadIdx.x;
    int st = gridDim.x * blockDim.x;
    const int total = POOL * D_MODEL * D_K;
    for (; i < total; i += st) {
        int n = i % D_K;
        int d = (i / D_K) % D_MODEL;
        int p = i / (D_K * D_MODEL);
        dst[((size_t)p * D_K + n) * D_MODEL + d] = f2bf(src[i]);
    }
}

// [D][D] f32 -> transposed [D][D] bf16: dst[n][d] = src[d][n]
__global__ void k_conv_sq_t(const float* __restrict__ src, u16* __restrict__ dst) {
    int i  = blockIdx.x * blockDim.x + threadIdx.x;
    int st = gridDim.x * blockDim.x;
    const int total = D_MODEL * D_MODEL;
    for (; i < total; i += st) {
        int n = i % D_MODEL;
        int d = i / D_MODEL;
        dst[(size_t)n * D_MODEL + d] = f2bf(src[i]);
    }
}

__global__ void k_zero(float* p) { if (threadIdx.x == 0 && blockIdx.x == 0) p[0] = 0.f; }

// ---------------------------------------------------------------- router
__global__ void k_router(const float* __restrict__ x, const float* __restrict__ u,
                         const float* __restrict__ r1w, const float* __restrict__ r1b,
                         const float* __restrict__ lng, const float* __restrict__ lnb,
                         const float* __restrict__ r2w, const float* __restrict__ r2b,
                         int* __restrict__ topidx) {
    __shared__ float pooled[D_MODEL];
    __shared__ float hbuf[POOL];
    __shared__ float zbuf[POOL];
    int b = blockIdx.x, t = threadIdx.x;
#pragma unroll
    for (int i = 0; i < 4; ++i) {
        int d = t + 256 * i;
        const float* xp = x + (size_t)b * SEQ * D_MODEL + d;
        float s = 0.f;
        for (int si = 0; si < SEQ; ++si) s += xp[(size_t)si * D_MODEL];
        pooled[d] = s * (1.0f / SEQ);
    }
    __syncthreads();
    if (t < POOL) {
        float a = r1b[t];
        for (int d = 0; d < D_MODEL; ++d) a += pooled[d] * r1w[d * POOL + t];
        hbuf[t] = a;
    }
    __syncthreads();
    if (t == 0) {
        float mu = 0.f;
        for (int p = 0; p < POOL; ++p) mu += hbuf[p];
        mu *= (1.0f / POOL);
        float var = 0.f;
        for (int p = 0; p < POOL; ++p) { float d = hbuf[p] - mu; var += d * d; }
        var *= (1.0f / POOL);
        float inv = rsqrtf(var + 1e-5f);
        for (int p = 0; p < POOL; ++p)
            hbuf[p] = fmaxf((hbuf[p] - mu) * inv * lng[p] + lnb[p], 0.f);
    }
    __syncthreads();
    if (t < POOL) {
        float a = r2b[t];
        for (int q = 0; q < POOL; ++q) a += hbuf[q] * r2w[q * POOL + t];
        float g = -logf(-logf(u[b * POOL + t]));
        zbuf[t] = (a + g) / TEMP;
    }
    __syncthreads();
    if (t == 0) {
        for (int h = 0; h < N_HEADS; ++h) {
            int best = 0; float bv = zbuf[0];
            for (int p = 1; p < POOL; ++p)
                if (zbuf[p] > bv) { bv = zbuf[p]; best = p; }
            topidx[b * N_HEADS + h] = best;
            zbuf[best] = -3.0e38f;
        }
    }
}

// ---------------------------------------------------------------- QKV GEMMs
// grid = B*H blocks, 256 threads (8 waves). Weights pre-transposed [dk][D].
// q,k stored row-major [s][dk]; v stored transposed [dk][s].
__global__ void k_qkv(const u16* __restrict__ xb,
                      const u16* __restrict__ wqt, const u16* __restrict__ wkt,
                      const u16* __restrict__ wvt,
                      const float* __restrict__ bq, const float* __restrict__ bk,
                      const float* __restrict__ bv,
                      const int* __restrict__ topidx, u16* __restrict__ qkv) {
    const int bh   = blockIdx.x;
    const int b    = bh >> 4;
    const int wave = threadIdx.x >> 5, lane = threadIdx.x & 31;
    const int lh   = lane >> 4, ln = lane & 15;
    const int idx  = topidx[bh];
    const size_t HSTRIDE = (size_t)BATCH * N_HEADS * SEQ * D_K;
    const u16*   Ws[3] = { wqt + (size_t)idx * D_MODEL * D_K,
                           wkt + (size_t)idx * D_MODEL * D_K,
                           wvt + (size_t)idx * D_MODEL * D_K };
    const float* Bs[3] = { bq + idx * D_K, bk + idx * D_K, bv + idx * D_K };
#pragma unroll
    for (int which = 0; which < 3; ++which) {
        u16* out = qkv + (size_t)which * HSTRIDE + (size_t)bh * SEQ * D_K;
        for (int pass = 0; pass < 2; ++pass) {
            int rt0 = wave * 4 + pass * 2;
            f32x8 acc[2][4];
#pragma unroll
            for (int i = 0; i < 2; ++i)
#pragma unroll
                for (int j = 0; j < 4; ++j)
#pragma unroll
                    for (int e = 0; e < 8; ++e) acc[i][j][e] = 0.f;
            for (int kt = 0; kt < D_MODEL / 32; ++kt) {
                bf16x16 a0 = load_a(xb + ((size_t)b * SEQ + rt0 * 16) * D_MODEL + kt * 32,
                                    D_MODEL, lane);
                bf16x16 a1 = load_a(xb + ((size_t)b * SEQ + rt0 * 16 + 16) * D_MODEL + kt * 32,
                                    D_MODEL, lane);
#pragma unroll
                for (int ct = 0; ct < 4; ++ct) {
                    // transposed weights: column n -> contiguous K (=d) values
                    bf16x16 bb = load_b_kcontig(Ws[which] + (size_t)(ct * 16) * D_MODEL + kt * 32,
                                                D_MODEL, lane);
                    acc[0][ct] = wmma_bf16(a0, bb, acc[0][ct]);
                    acc[1][ct] = wmma_bf16(a1, bb, acc[1][ct]);
                }
            }
            if (which < 2) {
                // q / k: row-major [s][dk]
#pragma unroll
                for (int rt = 0; rt < 2; ++rt)
#pragma unroll
                    for (int ct = 0; ct < 4; ++ct) {
                        int n = ct * 16 + ln;
                        float bias = Bs[which][n];
#pragma unroll
                        for (int g = 0; g < 8; ++g) {
                            int row = (rt0 + rt) * 16 + lh * 8 + g;
                            out[(size_t)row * D_K + n] = f2bf(acc[rt][ct][g] + bias);
                        }
                    }
            } else {
                // v: transposed [dk][s]; 8 consecutive rows per lane -> one b128 store
#pragma unroll
                for (int rt = 0; rt < 2; ++rt)
#pragma unroll
                    for (int ct = 0; ct < 4; ++ct) {
                        int n = ct * 16 + ln;
                        float bias = Bs[which][n];
                        int row0 = (rt0 + rt) * 16 + lh * 8;
                        uint4 pk;
                        pk.x = pack2bf(acc[rt][ct][0] + bias, acc[rt][ct][1] + bias);
                        pk.y = pack2bf(acc[rt][ct][2] + bias, acc[rt][ct][3] + bias);
                        pk.z = pack2bf(acc[rt][ct][4] + bias, acc[rt][ct][5] + bias);
                        pk.w = pack2bf(acc[rt][ct][6] + bias, acc[rt][ct][7] + bias);
                        *(uint4*)(out + (size_t)n * SEQ + row0) = pk;
                    }
            }
        }
    }
}

// ---------------------------------------------------------------- flash attention
// grid = B*H*4 blocks (128 query rows each), 128 threads (4 waves, 32 rows/wave).
__global__ void k_attn(const u16* __restrict__ qkv, u16* __restrict__ mh,
                       float* __restrict__ ent) {
    const size_t HSTRIDE = (size_t)BATCH * N_HEADS * SEQ * D_K;
    int bid = blockIdx.x;
    int bh = bid >> 2, rb = (bid & 3) * 128;
    int b = bh >> 4, h = bh & 15;
    int wave = threadIdx.x >> 5, lane = threadIdx.x & 31;
    int lh = lane >> 4, ln = lane & 15;
    const u16* qp = qkv + (size_t)bh * SEQ * D_K;
    const u16* kp = qp + HSTRIDE;          // [t][dk]
    const u16* vt = qp + 2 * HSTRIDE;      // [dk][t] (transposed)
    int r0 = rb + wave * 32;

    __shared__ u16 plds[4][32][32];

    bf16x16 qf[2][2];
#pragma unroll
    for (int rt = 0; rt < 2; ++rt)
#pragma unroll
        for (int kt = 0; kt < 2; ++kt)
            qf[rt][kt] = load_a(qp + (size_t)(r0 + rt * 16) * D_K + kt * 32, D_K, lane);

    f32x8 acc[2][4];
    float m_[2][8], l_[2][8], u_[2][8];
#pragma unroll
    for (int rt = 0; rt < 2; ++rt)
#pragma unroll
        for (int g = 0; g < 8; ++g) { m_[rt][g] = -3.0e38f; l_[rt][g] = 0.f; u_[rt][g] = 0.f; }
#pragma unroll
    for (int i = 0; i < 2; ++i)
#pragma unroll
        for (int j = 0; j < 4; ++j)
#pragma unroll
            for (int e = 0; e < 8; ++e) acc[i][j][e] = 0.f;

    for (int j = 0; j < SEQ / 32; ++j) {
        int j0 = j * 32;
        bf16x16 kb[2][2];
#pragma unroll
        for (int c2 = 0; c2 < 2; ++c2)
#pragma unroll
            for (int kt = 0; kt < 2; ++kt)
                kb[c2][kt] = load_b_kcontig(kp + (size_t)(j0 + c2 * 16) * D_K + kt * 32,
                                            D_K, lane);
        f32x8 sc[2][2];
#pragma unroll
        for (int rt = 0; rt < 2; ++rt)
#pragma unroll
            for (int c2 = 0; c2 < 2; ++c2) {
                f32x8 s;
#pragma unroll
                for (int e = 0; e < 8; ++e) s[e] = 0.f;
                s = wmma_bf16(qf[rt][0], kb[c2][0], s);
                s = wmma_bf16(qf[rt][1], kb[c2][1], s);
#pragma unroll
                for (int e = 0; e < 8; ++e) s[e] *= 0.125f;   // 1/sqrt(D_K)
                sc[rt][c2] = s;
            }
        // online softmax update (row stats across the 16-lane half)
#pragma unroll
        for (int rt = 0; rt < 2; ++rt) {
#pragma unroll
            for (int g = 0; g < 8; ++g) {
                float s0 = sc[rt][0][g], s1 = sc[rt][1][g];
                float v = fmaxf(s0, s1);
                for (int off = 8; off > 0; off >>= 1) v = fmaxf(v, __shfl_xor(v, off, 32));
                float mo = m_[rt][g];
                float mn = fmaxf(mo, v);
                float scal = __expf(mo - mn);
                float p0 = __expf(s0 - mn), p1 = __expf(s1 - mn);
                float rs = p0 + p1;
                float us = p0 * s0 + p1 * s1;
                for (int off = 8; off > 0; off >>= 1) {
                    rs += __shfl_xor(rs, off, 32);
                    us += __shfl_xor(us, off, 32);
                }
                l_[rt][g] = l_[rt][g] * scal + rs;
                u_[rt][g] = u_[rt][g] * scal + us;
                m_[rt][g] = mn;
#pragma unroll
                for (int ct = 0; ct < 4; ++ct) acc[rt][ct][g] *= scal;
                plds[wave][rt * 16 + lh * 8 + g][ln]      = f2bf(p0);
                plds[wave][rt * 16 + lh * 8 + g][16 + ln] = f2bf(p1);
            }
        }
        // V B-fragments from transposed v: column n -> contiguous keys
        bf16x16 vb[4];
#pragma unroll
        for (int ct = 0; ct < 4; ++ct)
            vb[ct] = load_b_kcontig(vt + (size_t)(ct * 16) * SEQ + j0, SEQ, lane);
        // P A-fragments from LDS (wave-private region, row stride 32)
#pragma unroll
        for (int rt = 0; rt < 2; ++rt) {
            FragAB pf;
            int ko = lh * 8;
            const u16* pb = &plds[wave][rt * 16 + ln][0];
            pf.q[0] = *(const uint4*)(pb + ko);
            pf.q[1] = *(const uint4*)(pb + ko + 16);
#pragma unroll
            for (int ct = 0; ct < 4; ++ct)
                acc[rt][ct] = wmma_bf16(pf.v, vb[ct], acc[rt][ct]);
        }
    }

    float hsum = 0.f;
#pragma unroll
    for (int rt = 0; rt < 2; ++rt)
#pragma unroll
        for (int g = 0; g < 8; ++g) {
            float l = l_[rt][g];
            float inv = 1.0f / l;
            int row = r0 + rt * 16 + lh * 8 + g;
#pragma unroll
            for (int ct = 0; ct < 4; ++ct) {
                int n = ct * 16 + ln;
                mh[((size_t)b * SEQ + row) * D_MODEL + h * D_K + n] = f2bf(acc[rt][ct][g] * inv);
            }
            hsum += m_[rt][g] + __logf(l) - u_[rt][g] * inv;
        }
    if (ln == 0) atomicAdd(ent, hsum);
}

// ---------------------------------------------------------------- output projection
// grid = 128 (M blocks) x 16 (N blocks) = 2048 blocks, 128 threads (4 waves).
// out_w pre-transposed: [n][d] bf16.
__global__ void k_outproj(const u16* __restrict__ mh, const u16* __restrict__ owt,
                          const float* __restrict__ ob, float* __restrict__ y) {
    int mb = blockIdx.x >> 4, nb = blockIdx.x & 15;
    int wave = threadIdx.x >> 5, lane = threadIdx.x & 31;
    int lh = lane >> 4, ln = lane & 15;
    int m0 = mb * 128 + wave * 32;
    int n0 = nb * 64;
    f32x8 acc[2][4];
#pragma unroll
    for (int i = 0; i < 2; ++i)
#pragma unroll
        for (int j = 0; j < 4; ++j)
#pragma unroll
            for (int e = 0; e < 8; ++e) acc[i][j][e] = 0.f;
    for (int kt = 0; kt < D_MODEL / 32; ++kt) {
        bf16x16 a0 = load_a(mh + (size_t)m0 * D_MODEL + kt * 32, D_MODEL, lane);
        bf16x16 a1 = load_a(mh + (size_t)(m0 + 16) * D_MODEL + kt * 32, D_MODEL, lane);
#pragma unroll
        for (int ct = 0; ct < 4; ++ct) {
            bf16x16 bb = load_b_kcontig(owt + (size_t)(n0 + ct * 16) * D_MODEL + kt * 32,
                                        D_MODEL, lane);
            acc[0][ct] = wmma_bf16(a0, bb, acc[0][ct]);
            acc[1][ct] = wmma_bf16(a1, bb, acc[1][ct]);
        }
    }
#pragma unroll
    for (int rt = 0; rt < 2; ++rt)
#pragma unroll
        for (int ct = 0; ct < 4; ++ct) {
            int n = n0 + ct * 16 + ln;
            float bias = ob[n];
#pragma unroll
            for (int g = 0; g < 8; ++g) {
                int row = m0 + rt * 16 + lh * 8 + g;
                y[(size_t)row * D_MODEL + n] = acc[rt][ct][g] + bias;
            }
        }
}

// ---------------------------------------------------------------- residual + LayerNorm
__global__ void k_final(const float* __restrict__ x, const float* __restrict__ y,
                        const float* __restrict__ g, const float* __restrict__ bb,
                        float* __restrict__ out) {
    __shared__ float red[256];
    __shared__ float sv[D_MODEL];
    int row = blockIdx.x, t = threadIdx.x;
    const float* xr = x + (size_t)row * D_MODEL;
    const float* yr = y + (size_t)row * D_MODEL;
    float ps = 0.f;
#pragma unroll
    for (int i = 0; i < 4; ++i) {
        int d = t + 256 * i;
        float s = xr[d] + yr[d];
        sv[d] = s;
        ps += s;
    }
    red[t] = ps;
    __syncthreads();
    for (int off = 128; off > 0; off >>= 1) {
        if (t < off) red[t] += red[t + off];
        __syncthreads();
    }
    float mu = red[0] * (1.0f / D_MODEL);
    __syncthreads();
    float pv = 0.f;
#pragma unroll
    for (int i = 0; i < 4; ++i) {
        int d = t + 256 * i;
        float dd = sv[d] - mu;
        pv += dd * dd;
    }
    red[t] = pv;
    __syncthreads();
    for (int off = 128; off > 0; off >>= 1) {
        if (t < off) red[t] += red[t + off];
        __syncthreads();
    }
    float inv = rsqrtf(red[0] * (1.0f / D_MODEL) + 1e-5f);
#pragma unroll
    for (int i = 0; i < 4; ++i) {
        int d = t + 256 * i;
        out[(size_t)row * D_MODEL + d] = (sv[d] - mu) * inv * g[d] + bb[d];
    }
}

__global__ void k_entropy_out(const float* __restrict__ ent, float* __restrict__ out) {
    if (threadIdx.x == 0 && blockIdx.x == 0)
        out[(size_t)BATCH * SEQ * D_MODEL] =
            ent[0] * (1.0f / ((float)BATCH * N_HEADS * SEQ));
}

// ---------------------------------------------------------------- launcher
extern "C" void kernel_launch(void* const* d_in, const int* in_sizes, int n_in,
                              void* d_out, int out_size, void* d_ws, size_t ws_size,
                              hipStream_t stream) {
    const float* x    = (const float*)d_in[0];
    const float* u    = (const float*)d_in[1];
    const float* Wq   = (const float*)d_in[2];
    const float* bq   = (const float*)d_in[3];
    const float* Wk   = (const float*)d_in[4];
    const float* bk   = (const float*)d_in[5];
    const float* Wv   = (const float*)d_in[6];
    const float* bv   = (const float*)d_in[7];
    const float* r1w  = (const float*)d_in[8];
    const float* r1b  = (const float*)d_in[9];
    const float* lng  = (const float*)d_in[10];
    const float* lnb  = (const float*)d_in[11];
    const float* r2w  = (const float*)d_in[12];
    const float* r2b  = (const float*)d_in[13];
    const float* outw = (const float*)d_in[14];
    const float* outb = (const float*)d_in[15];
    const float* ng   = (const float*)d_in[16];
    const float* nb   = (const float*)d_in[17];
    float* out = (float*)d_out;

    char* ws = (char*)d_ws;
    size_t o = 0;
    auto alloc = [&](size_t bytes) {
        size_t r = o;
        o += (bytes + 255) & ~(size_t)255;
        return r;
    };
    const size_t XBE = (size_t)BATCH * SEQ * D_MODEL;        // 16.7M
    const size_t WBE = (size_t)POOL * D_MODEL * D_K;         // 2.1M
    const size_t HBE = (size_t)BATCH * N_HEADS * SEQ * D_K;  // 16.7M

    u16*   xb    = (u16*)(ws + alloc(XBE * 2));
    u16*   wqt   = (u16*)(ws + alloc(WBE * 2));
    u16*   wkt   = (u16*)(ws + alloc(WBE * 2));
    u16*   wvt   = (u16*)(ws + alloc(WBE * 2));
    u16*   owt   = (u16*)(ws + alloc((size_t)D_MODEL * D_MODEL * 2));
    int*   tidx  = (int*)(ws + alloc((size_t)BATCH * N_HEADS * sizeof(int)));
    float* ent   = (float*)(ws + alloc(256));
    size_t o_qkv = alloc(3 * HBE * 2);                        // q,k (row-major), v (transposed)
    u16*   qkv   = (u16*)(ws + o_qkv);
    u16*   mh    = (u16*)(ws + alloc(XBE * 2));
    float* yb    = (float*)(ws + o_qkv);                      // alias: qkv dead by outproj

    k_zero<<<1, 1, 0, stream>>>(ent);
    k_f32_to_bf16<<<4096, 256, 0, stream>>>(x, xb, (int)XBE);
    k_conv_w_t<<<2048, 256, 0, stream>>>(Wq, wqt);
    k_conv_w_t<<<2048, 256, 0, stream>>>(Wk, wkt);
    k_conv_w_t<<<2048, 256, 0, stream>>>(Wv, wvt);
    k_conv_sq_t<<<2048, 256, 0, stream>>>(outw, owt);

    k_router<<<BATCH, 256, 0, stream>>>(x, u, r1w, r1b, lng, lnb, r2w, r2b, tidx);
    k_qkv<<<BATCH * N_HEADS, 256, 0, stream>>>(xb, wqt, wkt, wvt, bq, bk, bv, tidx, qkv);
    k_attn<<<BATCH * N_HEADS * 4, 128, 0, stream>>>(qkv, mh, ent);
    k_outproj<<<(SEQ * BATCH / 128) * (D_MODEL / 64), 128, 0, stream>>>(mh, owt, outb, yb);
    k_final<<<BATCH * SEQ, 256, 0, stream>>>(x, yb, ng, nb, out);
    k_entropy_out<<<1, 1, 0, stream>>>(ent, out);
}
